// RelFeatFusion_49555332661798
// MI455X (gfx1250) — compile-verified, matrix operand order checked by reading, not run on hardware
//
#include <hip/hip_runtime.h>
#include <hip/hip_bf16.h>
#include <math.h>

#define DEV __device__ __forceinline__

typedef unsigned short u16;
typedef __attribute__((ext_vector_type(16))) __bf16         v16bf;
typedef __attribute__((ext_vector_type(8)))  float          v8f;
typedef __attribute__((ext_vector_type(8)))  unsigned short us8;
typedef __attribute__((ext_vector_type(16))) unsigned short us16;

// ---------------- problem dimensions ----------------
constexpr int Dd   = 1936;            // embed dim
constexpr int Dp   = 1952;            // Dd padded to multiple of 32 (WMMA K)
constexpr int DF   = 2048;            // FFN dim
constexpr int D3   = 5808;            // 3*Dd
constexpr int D2   = 3872;            // 2*Dd (multiple of 32)
constexpr int NHn  = 8;
constexpr int HDn  = 242;             // head dim (not WMMA friendly -> VALU attn)
constexpr int Bimg = 512;
constexpr int Kobj = 32;
constexpr int Mtok = Bimg * Kobj;     // 16384 tokens, multiple of 256

// GEMM tiling
constexpr int ATILE = 256;            // block M tile
constexpr int BTILE = 128;            // block N tile
constexpr int LDH   = 40;             // LDS row pitch in halfs (80B: bank-conflict-free, 16B aligned)

DEV u16 cvt_bf16(float f) {
  union { float f; unsigned u; } v; v.f = f;
  unsigned r = v.u + 0x7FFFu + ((v.u >> 16) & 1u);
  return (u16)(r >> 16);
}

// ---------------- WMMA fragment load (16-bit A layout, 16x32) ----------------
// lanes 0-15 : row = lane,   K = 0..7 (v0..v3) and 16..23 (v4..v7)
// lanes 16-31: row = lane-16, K = 8..15 and 24..31
DEV v16bf ld_frag(const u16* tile, int row0, int lane) {
  const u16* p = tile + (row0 + (lane & 15)) * LDH + ((lane >> 4) << 3);
  us8 lo = *(const us8*)(p);
  us8 hi = *(const us8*)(p + 16);
  union { us16 u; v16bf b; } cc;
#pragma unroll
  for (int i = 0; i < 8; ++i) { cc.u[i] = lo[i]; cc.u[i + 8] = hi[i]; }
  return cc.b;
}

// ---------------- bf16 WMMA GEMM:  C[M,N] = A[M,K] * W[N,K]^T + bias ----------------
// 256x128 block tile, 8 wave32 waves each computing 64x64 (4x4 WMMA frags).
// K staged 32 at a time into double-buffered LDS via async global->LDS (ASYNCcnt).
// epi: 0=none, 1=relu, 2=tanh.  outF (f32, +col0F) and/or outB (bf16, pad-zeroed to npadB).
__global__ __launch_bounds__(256) void k_gemm(
    const u16* __restrict__ A, int lda,
    const u16* __restrict__ W, int ldw,
    const float* __restrict__ bias,
    float* __restrict__ outF, int ldoF, int col0F,
    u16* __restrict__ outB, int ldoB, int npadB,
    int N, int Kp, int epi)
{
  __shared__ u16 sA[2 * ATILE * LDH];   // 40 KB
  __shared__ u16 sB[2 * BTILE * LDH];   // 20 KB
  const int tx   = threadIdx.x;
  const int lane = tx & 31;
  const int w    = tx >> 5;
  const int wm   = (w >> 1) * 64;       // wave M offset within tile
  const int wn   = (w & 1) * 64;        // wave N offset within tile
  const int blockM = blockIdx.y * ATILE;
  const int blockN = blockIdx.x * BTILE;

  // staging: A -> thread stages one full 32-half row (4x16B); B -> half row (2x16B)
  const int arow = tx;                  // 0..255
  const int brow = tx >> 1;             // 0..127
  const int bco  = (tx & 1) << 4;       // half offset 0 / 16
  const int brG  = (blockN + brow < N) ? (blockN + brow) : (N - 1);  // clamp (guarded at store)

  const unsigned ldsA0 = (unsigned)(unsigned long long)(void*)&sA[arow * LDH];
  const unsigned ldsB0 = (unsigned)(unsigned long long)(void*)&sB[brow * LDH + bco];
  const unsigned bufAstride = (unsigned)(ATILE * LDH * sizeof(u16));
  const unsigned bufBstride = (unsigned)(BTILE * LDH * sizeof(u16));
  const unsigned long long gA0 = (unsigned long long)(A + (long)(blockM + arow) * lda);
  const unsigned long long gB0 = (unsigned long long)(W + (long)brG * ldw + bco);

  auto stage = [&](int buf, int k0) {
    const unsigned la = ldsA0 + (buf ? bufAstride : 0u);
    const unsigned lb = ldsB0 + (buf ? bufBstride : 0u);
    const unsigned long long ga = gA0 + (unsigned long long)(k0 * 2);
    const unsigned long long gb = gB0 + (unsigned long long)(k0 * 2);
    asm volatile(
        "global_load_async_to_lds_b128 %0, %2, off\n\t"
        "global_load_async_to_lds_b128 %0, %2, off offset:16\n\t"
        "global_load_async_to_lds_b128 %0, %2, off offset:32\n\t"
        "global_load_async_to_lds_b128 %0, %2, off offset:48\n\t"
        "global_load_async_to_lds_b128 %1, %3, off\n\t"
        "global_load_async_to_lds_b128 %1, %3, off offset:16"
        :: "v"(la), "v"(lb), "v"(ga), "v"(gb)
        : "memory");
  };

  stage(0, 0);

  const v8f vzero = {0.f,0.f,0.f,0.f,0.f,0.f,0.f,0.f};
  v8f acc[4][4];
#pragma unroll
  for (int i = 0; i < 4; ++i)
#pragma unroll
    for (int j = 0; j < 4; ++j) acc[i][j] = vzero;

  asm volatile("s_wait_asynccnt 0x0" ::: "memory");
  __syncthreads();

  const int nK = Kp >> 5;
  for (int kt = 0; kt < nK; ++kt) {
    const int cur = kt & 1;
    if (kt + 1 < nK) stage(cur ^ 1, (kt + 1) << 5);   // async prefetch into other buffer

    const u16* tA = &sA[cur * ATILE * LDH];
    const u16* tB = &sB[cur * BTILE * LDH];
    v16bf af[4], bfr[4];
#pragma unroll
    for (int mf = 0; mf < 4; ++mf) af[mf] = ld_frag(tA, wm + mf * 16, lane);
#pragma unroll
    for (int nf = 0; nf < 4; ++nf) bfr[nf] = ld_frag(tB, wn + nf * 16, lane);
#pragma unroll
    for (int mf = 0; mf < 4; ++mf)
#pragma unroll
      for (int nf = 0; nf < 4; ++nf)
        acc[mf][nf] = __builtin_amdgcn_wmma_f32_16x16x32_bf16(
            false, af[mf], false, bfr[nf], (short)0, acc[mf][nf], false, false);

    asm volatile("s_wait_asynccnt 0x0" ::: "memory");
    __syncthreads();
  }

  // epilogue: C 16x16 f32 layout -> VGPR i: lanes 0-15 M=i, lanes 16-31 M=i+8
  const int halfAdd = (lane >> 4) << 3;
  const int nlane   = lane & 15;
#pragma unroll
  for (int mf = 0; mf < 4; ++mf) {
#pragma unroll
    for (int nf = 0; nf < 4; ++nf) {
      const int gn = blockN + wn + nf * 16 + nlane;
      const int m0 = blockM + wm + mf * 16 + halfAdd;
      if (gn < N) {
        const float bv = bias ? bias[gn] : 0.f;
#pragma unroll
        for (int i = 0; i < 8; ++i) {
          float v = acc[mf][nf][i] + bv;
          if (epi == 1) v = fmaxf(v, 0.f);
          else if (epi == 2) v = tanhf(v);
          const long gm = (long)(m0 + i);
          if (outF) outF[gm * ldoF + col0F + gn] = v;
          if (outB) outB[gm * ldoB + gn] = cvt_bf16(v);
        }
      } else if (outB && gn < npadB) {
#pragma unroll
        for (int i = 0; i < 8; ++i) outB[(long)(m0 + i) * ldoB + gn] = (u16)0;
      }
    }
  }
}

// ---------------- LayerNorm (block per row), y = LN(x + res)*g + b ----------------
__global__ __launch_bounds__(256) void k_ln(
    const float* __restrict__ x, const float* __restrict__ res,
    const float* __restrict__ g, const float* __restrict__ bta,
    float* __restrict__ outF, u16* __restrict__ outB, int ldB)
{
  const int row = blockIdx.x;
  const int tx  = threadIdx.x;
  __shared__ float red[256];
  float vals[8];
  const float* px = x + (long)row * Dd;
  const float* pr = res + (long)row * Dd;
  float s = 0.f;
#pragma unroll
  for (int i = 0; i < 8; ++i) {
    const int c = tx + (i << 8);
    const float v = (c < Dd) ? (px[c] + pr[c]) : 0.f;
    vals[i] = v; s += v;
  }
  red[tx] = s; __syncthreads();
  for (int off = 128; off > 0; off >>= 1) { if (tx < off) red[tx] += red[tx + off]; __syncthreads(); }
  const float mean = red[0] * (1.f / (float)Dd);
  __syncthreads();
  float s2 = 0.f;
#pragma unroll
  for (int i = 0; i < 8; ++i) {
    const int c = tx + (i << 8);
    if (c < Dd) { const float d = vals[i] - mean; s2 += d * d; }
  }
  red[tx] = s2; __syncthreads();
  for (int off = 128; off > 0; off >>= 1) { if (tx < off) red[tx] += red[tx + off]; __syncthreads(); }
  const float inv = rsqrtf(red[0] * (1.f / (float)Dd) + 1e-5f);
#pragma unroll
  for (int i = 0; i < 8; ++i) {
    const int c = tx + (i << 8);
    if (c < Dd) {
      const float y = (vals[i] - mean) * inv * g[c] + bta[c];
      outF[(long)row * Dd + c] = y;
      if (outB) outB[(long)row * ldB + c] = cvt_bf16(y);
    } else if (outB && c < ldB) {
      outB[(long)row * ldB + c] = (u16)0;
    }
  }
}

// ---------------- encoder attention: 32 tokens per image ----------------
__global__ __launch_bounds__(256) void k_attn_enc(
    const float* __restrict__ qkv, float* __restrict__ out)
{
  const int b = blockIdx.x, h = blockIdx.y, tx = threadIdx.x;
  __shared__ float sc[32 * 32];
  const long rowBase = (long)b * Kobj;
  const int qo = h * HDn, ko = Dd + h * HDn, vo = 2 * Dd + h * HDn;
  const float scale = rsqrtf((float)HDn);
  for (int idx = tx; idx < 1024; idx += 256) {
    const int qi = idx >> 5, ki = idx & 31;
    const float* qp = qkv + (rowBase + qi) * (long)D3 + qo;
    const float* kp = qkv + (rowBase + ki) * (long)D3 + ko;
    float d = 0.f;
    for (int e = 0; e < HDn; ++e) d += qp[e] * kp[e];
    sc[idx] = d * scale;
  }
  __syncthreads();
  if (tx < 32) {
    float m = -1e30f;
    for (int k = 0; k < 32; ++k) m = fmaxf(m, sc[tx * 32 + k]);
    float s = 0.f;
    for (int k = 0; k < 32; ++k) { const float e = __expf(sc[tx*32+k] - m); sc[tx*32+k] = e; s += e; }
    const float r = 1.f / s;
    for (int k = 0; k < 32; ++k) sc[tx*32+k] *= r;
  }
  __syncthreads();
  for (int idx = tx; idx < 32 * HDn; idx += 256) {
    const int qi = idx / HDn, d = idx % HDn;
    float o = 0.f;
    for (int k = 0; k < 32; ++k)
      o += sc[qi * 32 + k] * qkv[(rowBase + k) * (long)D3 + vo + d];
    out[(rowBase + qi) * (long)Dd + h * HDn + d] = o;
  }
}

// ---------------- decoder attention: 512 tokens per object label ----------------
__global__ __launch_bounds__(256) void k_attn_dec(
    const float* __restrict__ qkv, float* __restrict__ out)
{
  const int q = blockIdx.x;   // query position 0..511
  const int h = blockIdx.y;   // head
  const int j = blockIdx.z;   // object label 0..31
  const int tx = threadIdx.x;
  __shared__ float sc[512];
  __shared__ float red[256];
  const long rowBase = (long)j * Bimg;
  const int qo = h * HDn, ko = Dd + h * HDn, vo = 2 * Dd + h * HDn;
  const float scale = rsqrtf((float)HDn);
  const float* qp = qkv + (rowBase + q) * (long)D3 + qo;
  float lmax = -1e30f;
  for (int k = tx; k < 512; k += 256) {
    const float* kp = qkv + (rowBase + k) * (long)D3 + ko;
    float d = 0.f;
    for (int e = 0; e < HDn; ++e) d += qp[e] * kp[e];
    d *= scale;
    sc[k] = d;
    lmax = fmaxf(lmax, d);
  }
  red[tx] = lmax; __syncthreads();
  for (int off = 128; off > 0; off >>= 1) { if (tx < off) red[tx] = fmaxf(red[tx], red[tx + off]); __syncthreads(); }
  const float m = red[0]; __syncthreads();
  float ls = 0.f;
  for (int k = tx; k < 512; k += 256) { const float e = __expf(sc[k] - m); sc[k] = e; ls += e; }
  red[tx] = ls; __syncthreads();
  for (int off = 128; off > 0; off >>= 1) { if (tx < off) red[tx] += red[tx + off]; __syncthreads(); }
  const float rs = 1.f / red[0];
  __syncthreads();
  for (int k = tx; k < 512; k += 256) sc[k] *= rs;
  __syncthreads();
  for (int d = tx; d < HDn; d += 256) {
    float o = 0.f;
    for (int k = 0; k < 512; ++k)
      o += sc[k] * qkv[(rowBase + k) * (long)D3 + vo + d];
    out[(rowBase + q) * (long)Dd + h * HDn + d] = o;
  }
}

// ---------------- elementwise helpers ----------------
__global__ void k_cvt(const float* __restrict__ src, u16* __restrict__ dst,
                      int cols, int ld, long total) {
  const long i = (long)blockIdx.x * 256 + threadIdx.x;
  if (i >= total) return;
  const long r = i / ld; const int c = (int)(i - r * ld);
  dst[i] = (c < cols) ? cvt_bf16(src[r * (long)cols + c]) : (u16)0;
}

__global__ void k_pos(float* __restrict__ pos) {
  const long i = (long)blockIdx.x * 256 + threadIdx.x;
  if (i >= (long)Bimg * Dd) return;
  const int p = (int)(i / Dd), c = (int)(i % Dd);
  const float li = 9.210340371976184f / 967.f; // log(1e4)/(num_ts-1)
  if (c < 968) pos[i] = __sinf((float)p * __expf(-li * (float)c));
  else         pos[i] = __cosf((float)p * __expf(-li * (float)(c - 968)));
}

// x[B,K,D] -> y[K,B,D]
__global__ void k_transpose(const float* __restrict__ x, float* __restrict__ y) {
  const long i = (long)blockIdx.x * 256 + threadIdx.x;
  if (i >= (long)Mtok * Dd) return;
  const long o = i / Dd; const int d = (int)(i - o * Dd);
  const int j = (int)(o >> 9); const int b = (int)(o & 511);
  y[i] = x[((long)b * Kobj + j) * Dd + d];
}

// produce y_bf and (y+pos)_bf, zero pads
__global__ void k_yprep(const float* __restrict__ y, const float* __restrict__ pos,
                        u16* __restrict__ yb, u16* __restrict__ ypb) {
  const long i = (long)blockIdx.x * 256 + threadIdx.x;
  if (i >= (long)Mtok * Dp) return;
  const long r = i / Dp; const int c = (int)(i - r * Dp);
  if (c < Dd) {
    const float v = y[r * (long)Dd + c];
    yb[i]  = cvt_bf16(v);
    ypb[i] = cvt_bf16(v + pos[(long)((int)(r & 511)) * Dd + c]);
  } else { yb[i] = 0; ypb[i] = 0; }
}

// y = t + f; refresh y_bf, (y+pos)_bf
__global__ void k_addupd(const float* __restrict__ t, const float* __restrict__ f,
                         const float* __restrict__ pos, float* __restrict__ y,
                         u16* __restrict__ yb, u16* __restrict__ ypb) {
  const long i = (long)blockIdx.x * 256 + threadIdx.x;
  if (i >= (long)Mtok * Dp) return;
  const long r = i / Dp; const int c = (int)(i - r * Dp);
  if (c < Dd) {
    const float v = t[r * (long)Dd + c] + f[r * (long)Dd + c];
    y[r * (long)Dd + c] = v;
    yb[i]  = cvt_bf16(v);
    ypb[i] = cvt_bf16(v + pos[(long)((int)(r & 511)) * Dd + c]);
  } else { yb[i] = 0; ypb[i] = 0; }
}

// fuse input [Mtok, 2D] bf16: cols [0,D)=shifted-back y (d0), cols [D,2D)=y
__global__ void k_fusein(const float* __restrict__ y, u16* __restrict__ fi) {
  const long i = (long)blockIdx.x * 256 + threadIdx.x;
  if (i >= (long)Mtok * D2) return;
  const long r = i / D2; const int c = (int)(i - r * D2);
  const int b = (int)(r & 511);
  float v;
  if (c < Dd) v = (b == 0) ? 0.f : y[(r - 1) * (long)Dd + c];
  else        v = y[r * (long)Dd + (c - Dd)];
  fi[i] = cvt_bf16(v);
}

// out[b*K+j][0:D]=y[j,b]; out[b*K+j][D:2D]=diff2[j,b]*y[j,max(b-1,0)]
__global__ void k_headout(const float* __restrict__ y, const float* __restrict__ diff2,
                          float* __restrict__ out) {
  const long i = (long)blockIdx.x * 256 + threadIdx.x;
  if (i >= (long)Mtok * D2) return;
  const long o = i / D2; const int c = (int)(i - o * D2);
  const int b = (int)(o >> 5); const int j = (int)(o & 31);
  const long rin = (long)j * Bimg + b;
  if (c < Dd) out[i] = y[rin * Dd + c];
  else {
    const int d = c - Dd;
    const long rm = (b == 0) ? rin : (rin - 1);
    out[i] = diff2[rin * Dd + d] * y[rm * Dd + d];
  }
}

// ---------------- host orchestration ----------------
extern "C" void kernel_launch(void* const* d_in, const int* in_sizes, int n_in,
                              void* d_out, int out_size, void* d_ws, size_t ws_size,
                              hipStream_t stream) {
  const float* features  = (const float*)d_in[0];
  const float* enc_qkv_w = (const float*)d_in[3];
  const float* enc_qkv_b = (const float*)d_in[4];
  const float* enc_out_w = (const float*)d_in[5];
  const float* enc_out_b = (const float*)d_in[6];
  const float* enc_ff1_w = (const float*)d_in[7];
  const float* enc_ff1_b = (const float*)d_in[8];
  const float* enc_ff2_w = (const float*)d_in[9];
  const float* enc_ff2_b = (const float*)d_in[10];
  const float* dec_qkv_w = (const float*)d_in[11];
  const float* dec_qkv_b = (const float*)d_in[12];
  const float* dec_out_w = (const float*)d_in[13];
  const float* dec_out_b = (const float*)d_in[14];
  const float* dec_ff1_w = (const float*)d_in[15];
  const float* dec_ff1_b = (const float*)d_in[16];
  const float* dec_ff2_w = (const float*)d_in[17];
  const float* dec_ff2_b = (const float*)d_in[18];
  const float* enc_ln1_g = (const float*)d_in[19];
  const float* enc_ln1_b = (const float*)d_in[20];
  const float* enc_ln2_g = (const float*)d_in[21];
  const float* enc_ln2_b = (const float*)d_in[22];
  const float* dec_ln_g  = (const float*)d_in[23];
  const float* dec_ln_b  = (const float*)d_in[24];
  const float* fuse_w    = (const float*)d_in[25];
  const float* fuse_b    = (const float*)d_in[26];
  const float* att1_w    = (const float*)d_in[27];
  const float* att1_b    = (const float*)d_in[28];
  const float* att2_w    = (const float*)d_in[29];
  const float* att2_b    = (const float*)d_in[30];

  size_t off = 0;
  auto alloc = [&](size_t bytes) -> void* {
    void* p = (char*)d_ws + off;
    off += (bytes + 255) & ~(size_t)255;
    return p;
  };

  // bf16 weight copies (reduction dim padded 1936->1952)
  u16* wEncQkv = (u16*)alloc((size_t)2 * D3 * Dp * 2);
  u16* wEncOut = (u16*)alloc((size_t)2 * Dd * Dp * 2);
  u16* wEncF1  = (u16*)alloc((size_t)2 * DF * Dp * 2);
  u16* wEncF2  = (u16*)alloc((size_t)2 * Dd * DF * 2);
  u16* wDecQkv = (u16*)alloc((size_t)2 * D3 * Dp * 2);
  u16* wDecOut = (u16*)alloc((size_t)2 * Dd * Dp * 2);
  u16* wDecF1  = (u16*)alloc((size_t)2 * DF * Dp * 2);
  u16* wDecF2  = (u16*)alloc((size_t)2 * Dd * DF * 2);
  u16* wFuse   = (u16*)alloc((size_t)Dd * D2 * 2);
  u16* wAtt1   = (u16*)alloc((size_t)Dd * Dp * 2);
  u16* wAtt2   = (u16*)alloc((size_t)Dd * Dp * 2);
  // activations
  float* pos      = (float*)alloc((size_t)Bimg * Dd * 4);
  u16*   x_bf     = (u16*)  alloc((size_t)Mtok * Dp * 2);
  float* xA       = (float*)alloc((size_t)Mtok * Dd * 4);
  float* xB       = (float*)alloc((size_t)Mtok * Dd * 4);
  float* qkv      = (float*)alloc((size_t)Mtok * D3 * 4);
  float* a_f32    = (float*)alloc((size_t)Mtok * Dd * 4);
  u16*   a_bf     = (u16*)  alloc((size_t)Mtok * Dp * 2);
  float* proj     = (float*)alloc((size_t)Mtok * Dd * 4);
  u16*   h_bf     = (u16*)  alloc((size_t)Mtok * DF * 2);
  float* f_f32    = (float*)alloc((size_t)Mtok * Dd * 4);
  float* y_f32    = (float*)alloc((size_t)Mtok * Dd * 4);
  u16*   y_bf     = (u16*)  alloc((size_t)Mtok * Dp * 2);
  u16*   ypos_bf  = (u16*)  alloc((size_t)Mtok * Dp * 2);
  float* t_f32    = (float*)alloc((size_t)Mtok * Dd * 4);
  u16*   fusein   = (u16*)  alloc((size_t)Mtok * D2 * 2);
  u16*   diff_bf  = (u16*)  alloc((size_t)Mtok * Dp * 2);
  u16*   t1_bf    = (u16*)  alloc((size_t)Mtok * Dp * 2);
  float* diff2    = (float*)alloc((size_t)Mtok * Dd * 4);

  auto nblk = [](long total) { return dim3((unsigned)((total + 255) / 256)); };
  auto cvt = [&](const float* src, u16* dst, int rows, int cols, int ld) {
    const long total = (long)rows * ld;
    k_cvt<<<nblk(total), 256, 0, stream>>>(src, dst, cols, ld, total);
  };
  auto gemm = [&](const u16* A, int lda, const u16* W, int ldw, const float* bias,
                  float* oF, int ldoF, int col0F, u16* oB, int ldoB, int npadB,
                  int N, int Kp, int epi) {
    dim3 g((unsigned)((N + BTILE - 1) / BTILE), Mtok / ATILE);
    k_gemm<<<g, 256, 0, stream>>>(A, lda, W, ldw, bias, oF, ldoF, col0F, oB, ldoB, npadB, N, Kp, epi);
  };

  // -------- weight conversion --------
  for (int l = 0; l < 2; ++l) {
    cvt(enc_qkv_w + (long)l * D3 * Dd, wEncQkv + (long)l * D3 * Dp, D3, Dd, Dp);
    cvt(enc_out_w + (long)l * Dd * Dd, wEncOut + (long)l * Dd * Dp, Dd, Dd, Dp);
    cvt(enc_ff1_w + (long)l * DF * Dd, wEncF1  + (long)l * DF * Dp, DF, Dd, Dp);
    cvt(enc_ff2_w + (long)l * Dd * DF, wEncF2  + (long)l * Dd * DF, Dd, DF, DF);
    cvt(dec_qkv_w + (long)l * D3 * Dd, wDecQkv + (long)l * D3 * Dp, D3, Dd, Dp);
    cvt(dec_out_w + (long)l * Dd * Dd, wDecOut + (long)l * Dd * Dp, Dd, Dd, Dp);
    cvt(dec_ff1_w + (long)l * DF * Dd, wDecF1  + (long)l * DF * Dp, DF, Dd, Dp);
    cvt(dec_ff2_w + (long)l * Dd * DF, wDecF2  + (long)l * Dd * DF, Dd, DF, DF);
  }
  cvt(fuse_w, wFuse, Dd, D2, D2);
  cvt(att1_w, wAtt1, Dd, Dd, Dp);
  cvt(att2_w, wAtt2, Dd, Dd, Dp);
  k_pos<<<nblk((long)Bimg * Dd), 256, 0, stream>>>(pos);

  // -------- encoder --------
  cvt(features, x_bf, Mtok, Dd, Dp);
  const float* cur = features;
  for (int l = 0; l < 2; ++l) {
    gemm(x_bf, Dp, wEncQkv + (long)l * D3 * Dp, Dp, enc_qkv_b + (long)l * D3,
         qkv, D3, 0, nullptr, 0, 0, D3, Dp, 0);
    k_attn_enc<<<dim3(Bimg, NHn), 256, 0, stream>>>(qkv, a_f32);
    cvt(a_f32, a_bf, Mtok, Dd, Dp);
    gemm(a_bf, Dp, wEncOut + (long)l * Dd * Dp, Dp, enc_out_b + (long)l * Dd,
         proj, Dd, 0, nullptr, 0, 0, Dd, Dp, 0);
    k_ln<<<Mtok, 256, 0, stream>>>(proj, cur, enc_ln1_g + (long)l * Dd,
                                   enc_ln1_b + (long)l * Dd, xA, x_bf, Dp);
    cur = xA;
    gemm(x_bf, Dp, wEncF1 + (long)l * DF * Dp, Dp, enc_ff1_b + (long)l * DF,
         nullptr, 0, 0, h_bf, DF, DF, DF, Dp, 1);
    gemm(h_bf, DF, wEncF2 + (long)l * Dd * DF, DF, enc_ff2_b + (long)l * Dd,
         f_f32, Dd, 0, nullptr, 0, 0, Dd, DF, 0);
    k_ln<<<Mtok, 256, 0, stream>>>(f_f32, cur, enc_ln2_g + (long)l * Dd,
                                   enc_ln2_b + (long)l * Dd, xB, x_bf, Dp);
    cur = xB;
    float* t = xA; xA = xB; xB = t;
  }

  // -------- decoder (layout [K, B, D]) --------
  k_transpose<<<nblk((long)Mtok * Dd), 256, 0, stream>>>(cur, y_f32);
  k_yprep<<<nblk((long)Mtok * Dp), 256, 0, stream>>>(y_f32, pos, y_bf, ypos_bf);
  for (int l = 0; l < 2; ++l) {
    const u16* wq = wDecQkv + (long)l * D3 * Dp;
    gemm(ypos_bf, Dp, wq, Dp, dec_qkv_b + (long)l * D3,
         qkv, D3, 0, nullptr, 0, 0, 2 * Dd, Dp, 0);                    // Q,K from y+pos
    gemm(y_bf, Dp, wq + (long)2 * Dd * Dp, Dp, dec_qkv_b + (long)l * D3 + 2 * Dd,
         qkv, D3, 2 * Dd, nullptr, 0, 0, Dd, Dp, 0);                   // V from y
    k_attn_dec<<<dim3(Bimg, NHn, Kobj), 256, 0, stream>>>(qkv, a_f32);
    cvt(a_f32, a_bf, Mtok, Dd, Dp);
    gemm(a_bf, Dp, wDecOut + (long)l * Dd * Dp, Dp, dec_out_b + (long)l * Dd,
         proj, Dd, 0, nullptr, 0, 0, Dd, Dp, 0);
    k_ln<<<Mtok, 256, 0, stream>>>(proj, y_f32, dec_ln_g + (long)l * Dd,
                                   dec_ln_b + (long)l * Dd, t_f32, x_bf, Dp);
    gemm(x_bf, Dp, wDecF1 + (long)l * DF * Dp, Dp, dec_ff1_b + (long)l * DF,
         nullptr, 0, 0, h_bf, DF, DF, DF, Dp, 1);
    gemm(h_bf, DF, wDecF2 + (long)l * Dd * DF, DF, dec_ff2_b + (long)l * Dd,
         f_f32, Dd, 0, nullptr, 0, 0, Dd, DF, 0);
    k_addupd<<<nblk((long)Mtok * Dp), 256, 0, stream>>>(t_f32, f_f32, pos,
                                                        y_f32, y_bf, ypos_bf);
  }

  // -------- fusion head --------
  k_fusein<<<nblk((long)Mtok * D2), 256, 0, stream>>>(y_f32, fusein);
  gemm(fusein, D2, wFuse, D2, fuse_b, nullptr, 0, 0, diff_bf, Dp, Dp, Dd, D2, 0);
  gemm(diff_bf, Dp, wAtt1, Dp, att1_b, nullptr, 0, 0, t1_bf, Dp, Dp, Dd, Dp, 2);
  gemm(t1_bf, Dp, wAtt2, Dp, att2_b, diff2, Dd, 0, nullptr, 0, 0, Dd, Dp, 2);
  k_headout<<<nblk((long)Mtok * D2), 256, 0, stream>>>(y_f32, diff2, (float*)d_out);
}
